// EncoderDNC_74234214744704
// MI455X (gfx1250) — matrix operand order repaired and checked by reading, not compiled, not run on hardware
//
#include <hip/hip_runtime.h>
#include <hip/hip_bf16.h>

#define T_STEPS 128
#define BATCH 64
#define IDIM_ 256
#define CDIM_ 512
#define G4 2048
#define R_HEADS 4
#define N_SLOTS 128
#define W_WORD 64
#define IFACE_ 471
#define IFACE_PAD 480
#define KOUT 768
#define EPSF 1e-6f
#define DELTAF 5e-6f
#define CLIPF 20.0f

// LDS row strides (halves), padded +8 halves (16B) to avoid bank conflicts:
// row stride 1040B -> bank step 4 -> lanes 0..15 hit distinct banks.
#define LDK (CDIM_ + 8)
#define LDKO (KOUT + 8)
#define LDKP (IDIM_ + 8)

typedef __attribute__((ext_vector_type(16))) _Float16 v16h;
typedef __attribute__((ext_vector_type(8))) float v8f;
typedef int v4i_ __attribute__((vector_size(16)));

// ---- async global->LDS staging (CDNA5 GLOBAL_LOAD_ASYNC_TO_LDS_B128, ASYNCcnt),
// with a safe synchronous fallback if the builtins are not declared.
#if defined(__has_builtin)
#if __has_builtin(__builtin_amdgcn_global_load_async_to_lds_b128) && \
    __has_builtin(__builtin_amdgcn_s_wait_asynccnt)
#define USE_ASYNC_LDS 1
#endif
#endif
#ifndef USE_ASYNC_LDS
#define USE_ASYNC_LDS 0
#endif

__device__ __forceinline__ void cp16(_Float16* l, const _Float16* g) {
#if USE_ASYNC_LDS
  __builtin_amdgcn_global_load_async_to_lds_b128(
      (v4i_*)g, (__attribute__((address_space(3))) v4i_*)l, 0, 0);
#else
  *(uint4*)l = *(const uint4*)g;
#endif
}
// Wait for this wave's async copies, then block barrier.
__device__ __forceinline__ void cp_commit() {
#if USE_ASYNC_LDS
  __builtin_amdgcn_s_wait_asynccnt(0);
#endif
  __syncthreads();
}

__device__ inline float sigf(float x) { return 1.f / (1.f + __expf(-x)); }
__device__ inline float softplusf(float x) { return (x > 20.f) ? x : log1pf(__expf(x)); }

// A fragment: 16x32 f16 tile, row-major source (rows = M, ld = row stride).
// ISA layout: lanes 0-15 -> M=lane, halves 0..7 = K 0..7, halves 8..15 = K 16..23
//             lanes 16-31 -> M=lane-16, halves 0..7 = K 8..15, halves 8..15 = K 24..31
__device__ __forceinline__ v16h frag_a(const _Float16* __restrict__ base, int ld) {
  int lane = threadIdx.x & 31;
  int r = lane & 15;
  int kb = (lane >> 4) << 3;
  const _Float16* p = base + (size_t)r * ld + kb;
  v16h f;
#pragma unroll
  for (int i = 0; i < 8; ++i) { f[i] = p[i]; f[i + 8] = p[i + 16]; }
  return f;
}

// B fragment: B[k][n] with source = row-major weight W[n][k] (ld = K stride).
// Layout: lanes 0-15 -> N=lane, halves 0..7 = K 0..7, 8..15 = K 16..23; lanes 16-31 -> N=lane-16, halves 0..7 = K 8..15, 8..15 = K 24..31.
__device__ __forceinline__ v16h frag_b(const _Float16* __restrict__ base, int ld) {
  int lane = threadIdx.x & 31;
  int n = lane & 15;
  int kb = (lane >> 4) << 3;
  const _Float16* p = base + (size_t)n * ld + kb;
  v16h f;
#pragma unroll
  for (int i = 0; i < 8; ++i) { f[i] = p[i]; f[i + 8] = p[i + 16]; }
  return f;
}

__device__ __forceinline__ v8f wmma16(v16h a, v16h b, v8f c) {
  return __builtin_amdgcn_wmma_f32_16x16x32_f16(false, a, false, b, (short)0, c,
                                                false, false);
}

// -------------------- conversion kernels --------------------
__global__ void k_f32_to_f16(const float* __restrict__ src, _Float16* __restrict__ dst, int n) {
  int i = blockIdx.x * 256 + threadIdx.x;
  if (i < n) dst[i] = (_Float16)src[i];
}

__global__ void k_wxi_f16(const float* __restrict__ src, _Float16* __restrict__ dst) {
  int i = blockIdx.x * 256 + threadIdx.x;
  if (i >= IFACE_PAD * CDIM_) return;
  int row = i / CDIM_;
  int col = i - row * CDIM_;
  dst[i] = (row < IFACE_) ? (_Float16)src[row * CDIM_ + col] : (_Float16)0.f;
}

// -------------------- time-parallel input projection --------------------
// Block = 4 waves over 4 consecutive M-tiles sharing one B tile in LDS.
// Output layout transposed per step: G[t][n][m] so the per-lane accumulator
// init in k_lstm0 reads 8 contiguous floats (2x b128).
__global__ void __launch_bounds__(128, 1)
k_gemm_pre(const _Float16* __restrict__ X, const _Float16* __restrict__ Wh,
           const float* __restrict__ b0, const float* __restrict__ b1,
           float* __restrict__ G) {
  __shared__ _Float16 lds_b[64 * LDKP];
  int n0 = blockIdx.y * 64;
  int tid = threadIdx.x;
  // stage B: rows n0..n0+63, K range 0..255 of the 512-wide weight rows
  for (int i = tid; i < 64 * (IDIM_ / 8); i += 128) {
    int row = i >> 5, v = i & 31;
    cp16(lds_b + row * LDKP + v * 8, Wh + (size_t)(n0 + row) * CDIM_ + v * 8);
  }
  cp_commit();
  int lane = tid & 31;
  int m0 = (blockIdx.x * 4 + (tid >> 5)) * 16;
  int col = lane & 15;
  int rb = (lane >> 4) << 3;
  v8f acc[4];
#pragma unroll
  for (int g = 0; g < 4; ++g) {
    float bv = b0[n0 + g * 16 + col] + b1[n0 + g * 16 + col];
#pragma unroll
    for (int r = 0; r < 8; ++r) acc[g][r] = bv;
  }
  const _Float16* Ab = X + (size_t)m0 * IDIM_;
#pragma unroll
  for (int kk = 0; kk < IDIM_; kk += 32) {
    v16h a = frag_a(Ab + kk, IDIM_);
#pragma unroll
    for (int g = 0; g < 4; ++g) {
      v16h bb = frag_b(lds_b + (size_t)(g * 16) * LDKP + kk, LDKP);
      acc[g] = wmma16(a, bb, acc[g]);
    }
  }
#pragma unroll
  for (int g = 0; g < 4; ++g)
#pragma unroll
    for (int r = 0; r < 8; ++r) {
      int gr = m0 + rb + r;                         // global row in 0..T*B
      G[((size_t)(gr >> 6) * G4 + n0 + g * 16 + col) * BATCH + (gr & 63)] = acc[g][r];
    }
}

// -------------------- LSTM layer 0 (recurrent GEMM + cell, fused gates) --------------------
// Block = 4 waves over the 4 batch M-tiles; B (4 gate tiles) and h both staged in LDS.
__global__ void __launch_bounds__(128, 1)
k_lstm0(const float* __restrict__ G, const _Float16* __restrict__ hin,
        const _Float16* __restrict__ Whh, float* __restrict__ cst,
        _Float16* __restrict__ hout, int t) {
  __shared__ _Float16 lds_b[64 * LDK];
  __shared__ _Float16 lds_a[64 * LDK];
  int c0c = blockIdx.x * 16;
  int tid = threadIdx.x;
  for (int i = tid; i < 64 * (CDIM_ / 8); i += 128) {
    int row = i >> 6, v = i & 63;
    int g = row >> 4, n = row & 15;
    cp16(lds_b + row * LDK + v * 8,
         Whh + (size_t)(g * CDIM_ + c0c + n) * CDIM_ + v * 8);
  }
  for (int i = tid; i < 64 * (CDIM_ / 8); i += 128) {
    int row = i >> 6, v = i & 63;
    cp16(lds_a + row * LDK + v * 8, hin + (size_t)row * CDIM_ + v * 8);
  }
  cp_commit();
  int lane = tid & 31;
  int m0 = (tid >> 5) * 16;
  int col = lane & 15;
  int rb = (lane >> 4) << 3;
  const float* Gt = G + (size_t)t * G4 * BATCH;
  v8f acc[4];
#pragma unroll
  for (int g = 0; g < 4; ++g) {
    const float* p = Gt + (size_t)(g * CDIM_ + c0c + col) * BATCH + m0 + rb;
#pragma unroll
    for (int r = 0; r < 8; ++r) acc[g][r] = p[r];   // contiguous -> 2x b128
  }
#pragma unroll
  for (int kk = 0; kk < CDIM_; kk += 32) {
    v16h a = frag_a(lds_a + (size_t)m0 * LDK + kk, LDK);
#pragma unroll
    for (int g = 0; g < 4; ++g) {
      v16h bb = frag_b(lds_b + (size_t)(g * 16) * LDK + kk, LDK);
      acc[g] = wmma16(a, bb, acc[g]);
    }
  }
#pragma unroll
  for (int r = 0; r < 8; ++r) {
    int row = m0 + rb + r;
    int cc = c0c + col;
    float iv = sigf(acc[0][r]);
    float fv = sigf(acc[1][r]);
    float gv = tanhf(acc[2][r]);
    float ov = sigf(acc[3][r]);
    float cold = cst[row * CDIM_ + cc];
    float cn = fv * cold + iv * gv;
    cst[row * CDIM_ + cc] = cn;
    hout[row * CDIM_ + cc] = (_Float16)(ov * tanhf(cn));
  }
}

// -------------------- LSTM layer 1 (two GEMMs + cell + clip) --------------------
// Two phases reusing the same 130KB LDS: phase 0 = (Wih, h0), phase 1 = (Whh, h1).
__global__ void __launch_bounds__(128, 1)
k_lstm1(const _Float16* __restrict__ h0c, const _Float16* __restrict__ h1p,
        const _Float16* __restrict__ Wih, const _Float16* __restrict__ Whh,
        const float* __restrict__ bih, const float* __restrict__ bhh,
        float* __restrict__ cst, _Float16* __restrict__ h1n,
        _Float16* __restrict__ outh) {
  __shared__ _Float16 lds_b[64 * LDK];
  __shared__ _Float16 lds_a[64 * LDK];
  int c0c = blockIdx.x * 16;
  int tid = threadIdx.x;
  int lane = tid & 31;
  int m0 = (tid >> 5) * 16;
  int col = lane & 15;
  int rb = (lane >> 4) << 3;
  v8f acc[4];
#pragma unroll
  for (int g = 0; g < 4; ++g) {
    float bv = bih[g * CDIM_ + c0c + col] + bhh[g * CDIM_ + c0c + col];
#pragma unroll
    for (int r = 0; r < 8; ++r) acc[g][r] = bv;
  }
#pragma unroll
  for (int phase = 0; phase < 2; ++phase) {
    const _Float16* W = phase ? Whh : Wih;
    const _Float16* H = phase ? h1p : h0c;
    for (int i = tid; i < 64 * (CDIM_ / 8); i += 128) {
      int row = i >> 6, v = i & 63;
      int g = row >> 4, n = row & 15;
      cp16(lds_b + row * LDK + v * 8,
           W + (size_t)(g * CDIM_ + c0c + n) * CDIM_ + v * 8);
    }
    for (int i = tid; i < 64 * (CDIM_ / 8); i += 128) {
      int row = i >> 6, v = i & 63;
      cp16(lds_a + row * LDK + v * 8, H + (size_t)row * CDIM_ + v * 8);
    }
    cp_commit();
#pragma unroll
    for (int kk = 0; kk < CDIM_; kk += 32) {
      v16h a = frag_a(lds_a + (size_t)m0 * LDK + kk, LDK);
#pragma unroll
      for (int g = 0; g < 4; ++g) {
        v16h bb = frag_b(lds_b + (size_t)(g * 16) * LDK + kk, LDK);
        acc[g] = wmma16(a, bb, acc[g]);
      }
    }
    __syncthreads();  // all waves done reading before phase-1 overwrite
  }
#pragma unroll
  for (int r = 0; r < 8; ++r) {
    int row = m0 + rb + r;
    int cc = c0c + col;
    float iv = sigf(acc[0][r]);
    float fv = sigf(acc[1][r]);
    float gv = tanhf(acc[2][r]);
    float ov = sigf(acc[3][r]);
    float cold = cst[row * CDIM_ + cc];
    float cn = fv * cold + iv * gv;
    float h = ov * tanhf(cn);
    cst[row * CDIM_ + cc] = cn;
    h1n[row * CDIM_ + cc] = (_Float16)h;
    float oc = fminf(fmaxf(h, -CLIPF), CLIPF);
    outh[row * CDIM_ + cc] = (_Float16)oc;
  }
}

// -------------------- interface projection xi = out @ Wxi.T + bxi --------------------
// Block = 4 waves over batch M-tiles; B tile (16 rows) + full out staged in LDS.
__global__ void __launch_bounds__(128, 1)
k_xi(const _Float16* __restrict__ outh, const _Float16* __restrict__ Wx,
     const float* __restrict__ bxi, float* __restrict__ xi) {
  __shared__ _Float16 lds_b[16 * LDK];
  __shared__ _Float16 lds_a[64 * LDK];
  int n0 = blockIdx.x * 16;
  int tid = threadIdx.x;
  for (int i = tid; i < 16 * (CDIM_ / 8); i += 128) {
    int row = i >> 6, v = i & 63;
    cp16(lds_b + row * LDK + v * 8, Wx + (size_t)(n0 + row) * CDIM_ + v * 8);
  }
  for (int i = tid; i < 64 * (CDIM_ / 8); i += 128) {
    int row = i >> 6, v = i & 63;
    cp16(lds_a + row * LDK + v * 8, outh + (size_t)row * CDIM_ + v * 8);
  }
  cp_commit();
  int lane = tid & 31;
  int m0 = (tid >> 5) * 16;
  int col = lane & 15;
  int rb = (lane >> 4) << 3;
  int n = n0 + col;
  v8f acc;
  float bv = (n < IFACE_) ? bxi[n] : 0.f;
#pragma unroll
  for (int r = 0; r < 8; ++r) acc[r] = bv;
#pragma unroll
  for (int kk = 0; kk < CDIM_; kk += 32) {
    v16h a = frag_a(lds_a + (size_t)m0 * LDK + kk, LDK);
    v16h bb = frag_b(lds_b + kk, LDK);
    acc = wmma16(a, bb, acc);
  }
  if (n < IFACE_) {
#pragma unroll
    for (int r = 0; r < 8; ++r)
      xi[(size_t)(m0 + rb + r) * IFACE_PAD + n] = acc[r];
  }
}

// -------------------- DNC memory controller: one workgroup per batch elem --------------------
__device__ inline float redsum(float* s, float v) {
  int tid = threadIdx.x;
  s[tid] = v;
  __syncthreads();
  for (int o = 64; o > 0; o >>= 1) {
    if (tid < o) s[tid] += s[tid + o];
    __syncthreads();
  }
  float r = s[0];
  __syncthreads();
  return r;
}
__device__ inline float redmax(float* s, float v) {
  int tid = threadIdx.x;
  s[tid] = v;
  __syncthreads();
  for (int o = 64; o > 0; o >>= 1) {
    if (tid < o) s[tid] = fmaxf(s[tid], s[tid + o]);
    __syncthreads();
  }
  float r = s[0];
  __syncthreads();
  return r;
}

__global__ void k_mem(const float* __restrict__ xi, float* __restrict__ mem,
                      float* __restrict__ link, float* __restrict__ prec,
                      float* __restrict__ rw, float* __restrict__ ww,
                      float* __restrict__ usage, _Float16* __restrict__ rv16) {
  __shared__ float s_link[N_SLOTS * N_SLOTS];
  __shared__ float s_rwp[R_HEADS * N_SLOTS];
  __shared__ float s_rwn[R_HEADS * N_SLOTS];
  __shared__ float s_u[N_SLOTS];
  __shared__ float s_ww[N_SLOTS];
  __shared__ float s_sorted[N_SLOTS];
  __shared__ float s_pexcl[N_SLOTS];
  __shared__ float s_norm[N_SLOTS];
  __shared__ float s_red[N_SLOTS];
  __shared__ float s_wkey[W_WORD];
  __shared__ float s_er[W_WORD];
  __shared__ float s_wv[W_WORD];
  __shared__ float s_rkey[R_HEADS * W_WORD];
  __shared__ float s_modes[12];
  __shared__ float s_fg[4];
  __shared__ float s_rstr[4];
  __shared__ float s_kn[4];
  __shared__ float s_sc[4];  // 0: alloc gate, 1: write gate, 2: write strength

  int b = blockIdx.x;
  int tid = threadIdx.x;
  const float* xb = xi + (size_t)b * IFACE_PAD;
  float* memb = mem + (size_t)b * N_SLOTS * W_WORD;
  float* linkb = link + (size_t)b * N_SLOTS * N_SLOTS;

  if (tid < 4) {
    s_fg[tid] = sigf(xb[453 + tid]);
    s_rstr[tid] = softplusf(xb[256 + tid]);
    float a0 = xb[459 + 3 * tid], a1 = xb[460 + 3 * tid], a2 = xb[461 + 3 * tid];
    float mx = fmaxf(a0, fmaxf(a1, a2));
    float e0 = __expf(a0 - mx), e1 = __expf(a1 - mx), e2 = __expf(a2 - mx);
    float s = e0 + e1 + e2;
    s_modes[3 * tid] = e0 / s;
    s_modes[3 * tid + 1] = e1 / s;
    s_modes[3 * tid + 2] = e2 / s;
  } else if (tid == 4) {
    s_sc[0] = sigf(xb[457]);
    s_sc[1] = sigf(xb[458]);
    s_sc[2] = softplusf(xb[324]);
  }
  if (tid < W_WORD) {
    s_wkey[tid] = tanhf(xb[260 + tid]);
    s_er[tid] = sigf(xb[325 + tid]);
    s_wv[tid] = tanhf(xb[389 + tid]);
  }
  for (int i = tid; i < R_HEADS * W_WORD; i += N_SLOTS) s_rkey[i] = tanhf(xb[i]);
  for (int i = tid; i < R_HEADS * N_SLOTS; i += N_SLOTS)
    s_rwp[i] = rw[(size_t)b * R_HEADS * N_SLOTS + i];
  __syncthreads();

  // ---- usage update
  float u = usage[b * N_SLOTS + tid];
  float wwp = ww[b * N_SLOTS + tid];
  u = u + (1.f - u) * wwp;
#pragma unroll
  for (int r = 0; r < 4; ++r) u *= (1.f - s_fg[r] * s_rwp[r * N_SLOTS + tid]);

  // ---- write-content weights (cosine softmax over slots; tid = slot)
  float wkn;
  {
    float v = (tid < W_WORD) ? s_wkey[tid] * s_wkey[tid] : 0.f;
    wkn = sqrtf(redsum(s_red, v)) + EPSF;
  }
  float ss = 0.f, dotw = 0.f;
  for (int w = 0; w < W_WORD; ++w) {
    float m = memb[tid * W_WORD + w];
    ss += m * m;
    dotw += m * s_wkey[w];
  }
  float sim = dotw / ((sqrtf(ss) + EPSF) * wkn) * s_sc[2];
  float mx = redmax(s_red, sim);
  float e = __expf(sim - mx);
  float wc = e / redsum(s_red, e);

  // ---- allocation via rank sort of usage
  float uu = DELTAF + (1.f - DELTAF) * u;
  s_u[tid] = uu;
  __syncthreads();
  int rank = 0;
  for (int j = 0; j < N_SLOTS; ++j) {
    float uj = s_u[j];
    rank += (uj < uu) || (uj == uu && j < tid);
  }
  s_sorted[rank] = uu;
  __syncthreads();
  if (tid == 0) {
    float p = 1.f;
    for (int k = 0; k < N_SLOTS; ++k) {
      s_pexcl[k] = p;
      p *= s_sorted[k];
    }
  }
  __syncthreads();
  float alloc = (1.f - uu) * s_pexcl[rank];
  float ag = s_sc[0], wg = s_sc[1];
  float wwn = wg * (ag * alloc + (1.f - ag) * wc);
  s_ww[tid] = wwn;
  usage[b * N_SLOTS + tid] = u;
  ww[b * N_SLOTS + tid] = wwn;
  float wwsum = redsum(s_red, wwn);

  // ---- memory write + new row norms
  ss = 0.f;
  for (int w = 0; w < W_WORD; ++w) {
    float m = memb[tid * W_WORD + w];
    m = m * (1.f - wwn * s_er[w]) + wwn * s_wv[w];
    memb[tid * W_WORD + w] = m;
    ss += m * m;
  }
  s_norm[tid] = sqrtf(ss) + EPSF;
  __syncthreads();

  // ---- link matrix update (tid = column j)
  float pr = prec[b * N_SLOTS + tid];
  for (int i = 0; i < N_SLOTS; ++i) {
    float l = linkb[i * N_SLOTS + tid];
    l = (1.f - s_ww[i] - wwn) * l + s_ww[i] * pr;
    if (i == tid) l = 0.f;
    s_link[i * N_SLOTS + tid] = l;
    linkb[i * N_SLOTS + tid] = l;
  }
  prec[b * N_SLOTS + tid] = (1.f - wwsum) * pr + wwn;

  if (tid < 4) {
    float s = 0.f;
    for (int w = 0; w < W_WORD; ++w) {
      float k = s_rkey[tid * W_WORD + w];
      s += k * k;
    }
    s_kn[tid] = sqrtf(s) + EPSF;
  }
  __syncthreads();

  // ---- read-content softmaxes (tid = slot)
  float rcv[4];
  for (int r = 0; r < 4; ++r) {
    float d = 0.f;
    for (int w = 0; w < W_WORD; ++w) d += s_rkey[r * W_WORD + w] * memb[tid * W_WORD + w];
    float si = d / (s_kn[r] * s_norm[tid]) * s_rstr[r];
    float m2 = redmax(s_red, si);
    float ee = __expf(si - m2);
    rcv[r] = ee / redsum(s_red, ee);
  }

  // ---- fwd/bwd temporal weights + new read weights
  for (int r = 0; r < 4; ++r) {
    float fwd = 0.f, bwd = 0.f;
    for (int m = 0; m < N_SLOTS; ++m) {
      float rp = s_rwp[r * N_SLOTS + m];
      fwd += rp * s_link[tid * N_SLOTS + m];
      bwd += rp * s_link[m * N_SLOTS + tid];
    }
    float rn = s_modes[3 * r] * bwd + s_modes[3 * r + 1] * rcv[r] + s_modes[3 * r + 2] * fwd;
    s_rwn[r * N_SLOTS + tid] = rn;
    rw[(size_t)b * R_HEADS * N_SLOTS + r * N_SLOTS + tid] = rn;
  }
  __syncthreads();

  // ---- read vectors (f16 for the output GEMM)
  for (int i = tid; i < R_HEADS * W_WORD; i += N_SLOTS) {
    int r = i >> 6, w = i & 63;
    float a = 0.f;
    for (int n2 = 0; n2 < N_SLOTS; ++n2) a += s_rwn[r * N_SLOTS + n2] * memb[n2 * W_WORD + w];
    rv16[(size_t)b * (R_HEADS * W_WORD) + i] = (_Float16)a;
  }
}

// -------------------- output GEMM y = [out, read_vecs] @ Wout.T + bout --------------------
// Block = 4 waves over batch M-tiles; Wout tile (64 rows x 768) staged in LDS.
__global__ void __launch_bounds__(128, 1)
k_out(const _Float16* __restrict__ outh, const _Float16* __restrict__ rvh,
      const _Float16* __restrict__ Wo, const float* __restrict__ bo,
      float* __restrict__ y, int t) {
  __shared__ _Float16 lds_b[64 * LDKO];
  int n0 = blockIdx.x * 64;
  int tid = threadIdx.x;
  for (int i = tid; i < 64 * (KOUT / 8); i += 128) {
    int row = i / (KOUT / 8), v = i % (KOUT / 8);
    cp16(lds_b + row * LDKO + v * 8, Wo + (size_t)(n0 + row) * KOUT + v * 8);
  }
  cp_commit();
  int lane = tid & 31;
  int m0 = (tid >> 5) * 16;
  int col = lane & 15;
  int rb = (lane >> 4) << 3;
  v8f acc[4];
#pragma unroll
  for (int g = 0; g < 4; ++g) {
    float bv = bo[n0 + g * 16 + col];
#pragma unroll
    for (int r = 0; r < 8; ++r) acc[g][r] = bv;
  }
  const _Float16* A0 = outh + (size_t)m0 * CDIM_;
  const _Float16* A1 = rvh + (size_t)m0 * (R_HEADS * W_WORD);
#pragma unroll
  for (int kk = 0; kk < KOUT; kk += 32) {
    v16h a = (kk < CDIM_) ? frag_a(A0 + kk, CDIM_)
                          : frag_a(A1 + (kk - CDIM_), R_HEADS * W_WORD);
#pragma unroll
    for (int g = 0; g < 4; ++g) {
      v16h bb = frag_b(lds_b + (size_t)(g * 16) * LDKO + kk, LDKO);
      acc[g] = wmma16(a, bb, acc[g]);
    }
  }
#pragma unroll
  for (int g = 0; g < 4; ++g)
#pragma unroll
    for (int r = 0; r < 8; ++r)
      y[((size_t)t * BATCH + m0 + rb + r) * IDIM_ + n0 + g * 16 + col] = acc[g][r];
}

// -------------------- final hidden gather --------------------
__global__ void k_hid(const float* __restrict__ y, const int* __restrict__ lens,
                      float* __restrict__ hid) {
  int b = blockIdx.x;
  int c = threadIdx.x;
  int t = lens[b] - 1;
  hid[b * IDIM_ + c] = y[((size_t)t * BATCH + b) * IDIM_ + c];
}

extern "C" void kernel_launch(void* const* d_in, const int* in_sizes, int n_in,
                              void* d_out, int out_size, void* d_ws, size_t ws_size,
                              hipStream_t stream) {
  (void)in_sizes; (void)n_in; (void)out_size; (void)ws_size;
  const float* embs = (const float*)d_in[0];
  const int* lens = (const int*)d_in[1];
  const float* Wih0 = (const float*)d_in[2];
  const float* Whh0 = (const float*)d_in[3];
  const float* bih0 = (const float*)d_in[4];
  const float* bhh0 = (const float*)d_in[5];
  const float* Wih1 = (const float*)d_in[6];
  const float* Whh1 = (const float*)d_in[7];
  const float* bih1 = (const float*)d_in[8];
  const float* bhh1 = (const float*)d_in[9];
  const float* Wxi = (const float*)d_in[10];
  const float* bxi = (const float*)d_in[11];
  const float* Wout = (const float*)d_in[12];
  const float* bout = (const float*)d_in[13];

  char* ws = (char*)d_ws;
  size_t off = 0;
  auto take = [&](size_t bytes) -> char* {
    char* p = ws + off;
    off += (bytes + 255) & ~(size_t)255;
    return p;
  };
  _Float16* embs16 = (_Float16*)take((size_t)T_STEPS * BATCH * IDIM_ * 2);
  _Float16* Wih0h = (_Float16*)take((size_t)G4 * CDIM_ * 2);
  _Float16* Whh0h = (_Float16*)take((size_t)G4 * CDIM_ * 2);
  _Float16* Wih1h = (_Float16*)take((size_t)G4 * CDIM_ * 2);
  _Float16* Whh1h = (_Float16*)take((size_t)G4 * CDIM_ * 2);
  _Float16* Wxih = (_Float16*)take((size_t)IFACE_PAD * CDIM_ * 2);
  _Float16* Wouth = (_Float16*)take((size_t)IDIM_ * KOUT * 2);
  float* G0 = (float*)take((size_t)T_STEPS * BATCH * G4 * 4);
  // zero-initialized state block (contiguous)
  char* state0 = ws + off;
  _Float16* h016 = (_Float16*)take((size_t)2 * BATCH * CDIM_ * 2);
  _Float16* h116 = (_Float16*)take((size_t)2 * BATCH * CDIM_ * 2);
  _Float16* out16 = (_Float16*)take((size_t)BATCH * CDIM_ * 2);
  _Float16* rv16 = (_Float16*)take((size_t)BATCH * R_HEADS * W_WORD * 2);
  float* c0 = (float*)take((size_t)BATCH * CDIM_ * 4);
  float* c1 = (float*)take((size_t)BATCH * CDIM_ * 4);
  float* mem = (float*)take((size_t)BATCH * N_SLOTS * W_WORD * 4);
  float* link = (float*)take((size_t)BATCH * N_SLOTS * N_SLOTS * 4);
  float* prec = (float*)take((size_t)BATCH * N_SLOTS * 4);
  float* rw = (float*)take((size_t)BATCH * R_HEADS * N_SLOTS * 4);
  float* wwv = (float*)take((size_t)BATCH * N_SLOTS * 4);
  float* usage = (float*)take((size_t)BATCH * N_SLOTS * 4);
  float* xiv = (float*)take((size_t)BATCH * IFACE_PAD * 4);
  size_t state_bytes = (size_t)((ws + off) - state0);
  (void)hipMemsetAsync(state0, 0, state_bytes, stream);

  int n;
  n = T_STEPS * BATCH * IDIM_;
  k_f32_to_f16<<<(n + 255) / 256, 256, 0, stream>>>(embs, embs16, n);
  n = G4 * CDIM_;
  k_f32_to_f16<<<(n + 255) / 256, 256, 0, stream>>>(Wih0, Wih0h, n);
  k_f32_to_f16<<<(n + 255) / 256, 256, 0, stream>>>(Whh0, Whh0h, n);
  k_f32_to_f16<<<(n + 255) / 256, 256, 0, stream>>>(Wih1, Wih1h, n);
  k_f32_to_f16<<<(n + 255) / 256, 256, 0, stream>>>(Whh1, Whh1h, n);
  k_wxi_f16<<<(IFACE_PAD * CDIM_ + 255) / 256, 256, 0, stream>>>(Wxi, Wxih);
  n = IDIM_ * KOUT;
  k_f32_to_f16<<<(n + 255) / 256, 256, 0, stream>>>(Wout, Wouth, n);

  // time-parallel input projection: (T*B) x 2048 x 256
  k_gemm_pre<<<dim3(T_STEPS * BATCH / 64, G4 / 64), 128, 0, stream>>>(embs16, Wih0h, bih0,
                                                                      bhh0, G0);

  float* y = (float*)d_out;
  for (int t = 0; t < T_STEPS; ++t) {
    _Float16* h0in = h016 + (size_t)(t & 1) * BATCH * CDIM_;
    _Float16* h0out = h016 + (size_t)((t + 1) & 1) * BATCH * CDIM_;
    _Float16* h1in = h116 + (size_t)(t & 1) * BATCH * CDIM_;
    _Float16* h1out = h116 + (size_t)((t + 1) & 1) * BATCH * CDIM_;
    k_lstm0<<<CDIM_ / 16, 128, 0, stream>>>(G0, h0in, Whh0h, c0, h0out, t);
    k_lstm1<<<CDIM_ / 16, 128, 0, stream>>>(h0out, h1in, Wih1h, Whh1h, bih1, bhh1, c1,
                                            h1out, out16);
    k_xi<<<IFACE_PAD / 16, 128, 0, stream>>>(out16, Wxih, bxi, xiv);
    k_mem<<<BATCH, N_SLOTS, 0, stream>>>(xiv, mem, link, prec, rw, wwv, usage, rv16);
    k_out<<<IDIM_ / 64, 128, 0, stream>>>(out16, rv16, Wouth, bout, y, t);
  }
  k_hid<<<BATCH, IDIM_, 0, stream>>>(y, lens, y + (size_t)T_STEPS * BATCH * IDIM_);
}
